// GIN_65240553226750
// MI455X (gfx1250) — matrix operand order repaired and checked by reading, not compiled, not run on hardware
//
#include <hip/hip_runtime.h>

// ---------------------------------------------------------------------------
// GIN forward for MI455X (gfx1250, wave32).
//  - All GEMMs: v_wmma_f32_16x16x32_bf16, f32 accumulation.
//  - adj is binary {0,1} -> exact in bf16; f32->bf16 conversion during LDS
//    staging (no extra HBM pass).
//  - LDS tiles stored in WMMA *fragment order* (lane-swizzled): each operand
//    fragment is one contiguous 32B LDS read (2x ds_load_b128).
//  - agg kernel: double-buffered K pipeline — stage tile k+1 while computing
//    tile k, one barrier per iteration (global latency hidden behind WMMA).
//  - 256 threads = 8 waves per block; wave (r,g) owns rows r*16..r*16+15 and
//    column half g of the output panel.
// ---------------------------------------------------------------------------

typedef __attribute__((ext_vector_type(16))) __bf16 v16bf;
typedef __attribute__((ext_vector_type(2)))  __bf16 v2bf;
typedef __attribute__((ext_vector_type(8)))  float  v8f;

#define NNODES 1024
#define BATCH  32

static __device__ __forceinline__ __bf16 to_bf16(float f) { return (__bf16)f; }

static __device__ __forceinline__ v2bf pk_bf16(float x, float y) {
    v2bf r; r[0] = (__bf16)x; r[1] = (__bf16)y; return r;
}

// A-operand swizzle (16x32 bf16 tile per strip):
//   lane = half*16 + m ; element idx: e<8 <-> K=half*8+e, e>=8 <-> K=16+half*8+e
// For K-coordinate kc (0..31): half=(kc>>3)&1, idx=(kc&7)+((kc>>4)<<3).
// B-operand swizzle (32x16 bf16 tile): lane=((k>>3)&1)*16+(n&15), idx=(k&7)+((k>>4)<<3).

// ---------------------------------------------------------------------------
// Z[b,i,f] = H[b,i,f] + sum_j adj[b,i,j] * H[b,j,f]      (eps = 0)
// ---------------------------------------------------------------------------
template <int F>
__global__ __launch_bounds__(256)
void agg_gemm_kernel(const float* __restrict__ adj,   // [B,N,N]
                     const float* __restrict__ H,     // [B,N,F]
                     float* __restrict__ Z)           // [B,N,F]
{
    constexpr int MT = 64;        // output rows per block
    constexpr int KT = 32;        // K chunk per WMMA
    constexpr int NT = F / 32;    // 16-wide col tiles per wave
    constexpr int NSTRIP = MT / 16;
    constexpr int NK = NNODES / KT;

    // double-buffered fragment-order tiles
    __shared__ alignas(32) __bf16 Asw[2][NSTRIP * 32 * 16];
    __shared__ alignas(32) __bf16 Bsw[2][(F / 16) * 32 * 16];

    const int b       = blockIdx.y;
    const int rowBase = blockIdx.x * MT;
    const float* adjB = adj + (size_t)b * NNODES * NNODES + (size_t)rowBase * NNODES;
    const float* HB   = H   + (size_t)b * NNODES * F;
    float*       ZB   = Z   + (size_t)b * NNODES * F + (size_t)rowBase * F;

    const int tid  = threadIdx.x;
    const int wave = tid >> 5;
    const int lane = tid & 31;
    const int half = lane >> 4;
    const int l16  = lane & 15;
    const int wRow = (wave & 3) * 16;   // row strip base
    const int wGrp = wave >> 2;         // column half (0/1)

    // stage one K-tile (adj + features) into buffer `bsel`
    auto stage = [&](int bsel, int kk) {
        // A: 64x32 adj, float2 loads, paired bf16 stores (4 iters, uniform)
        for (int e = tid; e < MT * KT / 2; e += 256) {
            const int r = e >> 4;
            const int c = (e & 15) << 1;                     // even column
            const float2 v = *reinterpret_cast<const float2*>(
                adjB + (size_t)r * NNODES + kk + c);
            const int hf    = (c >> 3) & 1;
            const int idx   = (c & 7) + ((c >> 4) << 3);     // pair idx,idx+1
            const int lw    = hf * 16 + (r & 15);
            const int strip = r >> 4;
            *reinterpret_cast<v2bf*>(&Asw[bsel][((strip * 32 + lw) << 4) + idx]) =
                pk_bf16(v.x, v.y);
        }
        // B: 32xF features, paired along K (F/16 iters, uniform)
        for (int e = tid; e < (KT / 2) * F; e += 256) {
            const int kp = e / F;
            const int c  = e % F;
            const int k0 = kp << 1;                          // even K
            const float x0 = HB[(size_t)(kk + k0)     * F + c];
            const float x1 = HB[(size_t)(kk + k0 + 1) * F + c];
            const int lw   = (((k0 >> 3) & 1) << 4) + (c & 15);
            const int tile = c >> 4;
            const int idx  = (k0 & 7) + ((k0 >> 4) << 3);    // pair idx,idx+1
            *reinterpret_cast<v2bf*>(&Bsw[bsel][((tile * 32 + lw) << 4) + idx]) =
                pk_bf16(x0, x1);
        }
    };

    v8f acc[NT] = {};

    stage(0, 0);
    __syncthreads();

    for (int i = 0; i < NK; ++i) {
        const int cur = i & 1;
        // prefetch next tile into the other buffer (scalar-uniform condition;
        // overlaps its global loads with this iteration's WMMAs)
        if (i + 1 < NK) stage(cur ^ 1, (i + 1) * KT);

        // fragment loads: one contiguous 32B read per operand
        const v16bf af = *reinterpret_cast<const v16bf*>(
            &Asw[cur][((wave & 3) * 32 + lane) << 4]);
        #pragma unroll
        for (int t = 0; t < NT; ++t) {
            const int tIdx = wGrp * (F / 32) + t;
            const v16bf bfv = *reinterpret_cast<const v16bf*>(
                &Bsw[cur][(tIdx * 32 + lane) << 4]);
            acc[t] = __builtin_amdgcn_wmma_f32_16x16x32_bf16(
                false, af, false, bfv, (short)0, acc[t], false, false);
        }
        __syncthreads();   // single barrier per K-step
    }

    // epilogue: add self term (f32, full precision) and store f32
    #pragma unroll
    for (int t = 0; t < NT; ++t) {
        const int n = (wGrp * (F / 32) + t) * 16 + l16;
        #pragma unroll
        for (int r = 0; r < 8; ++r) {
            const int row = wRow + half * 8 + r;     // C/D layout: M = half*8 + r
            ZB[(size_t)row * F + n] =
                HB[(size_t)(rowBase + row) * F + n] + acc[t][r];
        }
    }
}

// ---------------------------------------------------------------------------
// Hout = relu(Z @ W1 + b1) @ W2 + b2, * mask    (both GEMMs fused)
// All panels staged once in fragment-order LDS.
// ---------------------------------------------------------------------------
template <int FIN, int HMID, int HOUT>
__global__ __launch_bounds__(256)
void mlp_gemm_kernel(const float* __restrict__ Zin,   // [B,N,FIN]
                     const float* __restrict__ W1,    // [FIN,HMID]
                     const float* __restrict__ b1,    // [HMID]
                     const float* __restrict__ W2,    // [HMID,HOUT]
                     const float* __restrict__ b2,    // [HOUT]
                     const float* __restrict__ mask,  // [B,N,1]
                     float* __restrict__ Hout)        // [B,N,HOUT]
{
    constexpr int MT   = 64;
    constexpr int NSTRIP = MT / 16;       // 4
    constexpr int KC1  = FIN / 32;        // stage-1 K chunks
    constexpr int KC2  = HMID / 32;       // stage-2 K chunks
    constexpr int NT1  = HMID / 32;       // col tiles per wave, stage 1
    constexpr int NT2  = HOUT / 32;       // col tiles per wave, stage 2

    __shared__ alignas(32) __bf16 Zsw [KC1 * NSTRIP      * 32 * 16];
    __shared__ alignas(32) __bf16 W1sw[KC1 * (HMID / 16) * 32 * 16];
    __shared__ alignas(32) __bf16 W2sw[KC2 * (HOUT / 16) * 32 * 16];
    __shared__ alignas(32) __bf16 Tsw [KC2 * NSTRIP      * 32 * 16];

    const int b       = blockIdx.y;
    const int rowBase = blockIdx.x * MT;
    const float* Zb = Zin  + ((size_t)b * NNODES + rowBase) * FIN;
    float*       Hb = Hout + ((size_t)b * NNODES + rowBase) * HOUT;

    const int tid  = threadIdx.x;
    const int wave = tid >> 5;
    const int lane = tid & 31;
    const int half = lane >> 4;
    const int l16  = lane & 15;
    const int wRow = (wave & 3) * 16;
    const int wGrp = wave >> 2;      // 0..1

    // ---- stage Z panel (A-operand): float2 loads, paired bf16 stores ----
    for (int e = tid; e < MT * FIN / 2; e += 256) {
        const int r = e / (FIN / 2);
        const int c = (e % (FIN / 2)) << 1;                  // even
        const float2 v = *reinterpret_cast<const float2*>(Zb + (size_t)r * FIN + c);
        const int KC  = c >> 5;
        const int kc  = c & 31;
        const int hf  = (kc >> 3) & 1;
        const int idx = (kc & 7) + ((kc >> 4) << 3);
        const int lw  = hf * 16 + (r & 15);
        const int strip = r >> 4;
        *reinterpret_cast<v2bf*>(
            &Zsw[(((KC * NSTRIP + strip) * 32 + lw) << 4) + idx]) = pk_bf16(v.x, v.y);
    }
    // ---- stage W1 panel (B-operand): paired along K ----
    for (int e = tid; e < FIN * HMID / 2; e += 256) {
        const int k0 = (e / HMID) << 1;
        const int c  = e % HMID;
        const float x0 = W1[(size_t)k0       * HMID + c];
        const float x1 = W1[(size_t)(k0 + 1) * HMID + c];
        const int KC   = k0 >> 5;
        const int kl   = k0 & 31;
        const int lw   = (((kl >> 3) & 1) << 4) + (c & 15);
        const int tile = c >> 4;
        const int idx  = (kl & 7) + ((kl >> 4) << 3);
        *reinterpret_cast<v2bf*>(
            &W1sw[(((KC * (HMID / 16) + tile) * 32 + lw) << 4) + idx]) = pk_bf16(x0, x1);
    }
    // ---- stage W2 panel (B-operand): paired along K ----
    for (int e = tid; e < HMID * HOUT / 2; e += 256) {
        const int k0 = (e / HOUT) << 1;
        const int c  = e % HOUT;
        const float x0 = W2[(size_t)k0       * HOUT + c];
        const float x1 = W2[(size_t)(k0 + 1) * HOUT + c];
        const int KC   = k0 >> 5;
        const int kl   = k0 & 31;
        const int lw   = (((kl >> 3) & 1) << 4) + (c & 15);
        const int tile = c >> 4;
        const int idx  = (kl & 7) + ((kl >> 4) << 3);
        *reinterpret_cast<v2bf*>(
            &W2sw[(((KC * (HOUT / 16) + tile) * 32 + lw) << 4) + idx]) = pk_bf16(x0, x1);
    }
    __syncthreads();

    // ---- stage 1: T = relu(Z @ W1 + b1) ----
    {
        v8f acc[NT1] = {};
        for (int kc = 0; kc < KC1; ++kc) {
            const v16bf af = *reinterpret_cast<const v16bf*>(
                &Zsw[((kc * NSTRIP + (wave & 3)) * 32 + lane) << 4]);
            #pragma unroll
            for (int t = 0; t < NT1; ++t) {
                const int tIdx = wGrp * (HMID / 32) + t;
                const v16bf bfv = *reinterpret_cast<const v16bf*>(
                    &W1sw[((kc * (HMID / 16) + tIdx) * 32 + lane) << 4]);
                acc[t] = __builtin_amdgcn_wmma_f32_16x16x32_bf16(
                    false, af, false, bfv, (short)0, acc[t], false, false);
            }
        }
        // bias + relu -> Tsw (A-operand fragment order for stage 2)
        #pragma unroll
        for (int t = 0; t < NT1; ++t) {
            const int n = (wGrp * (HMID / 32) + t) * 16 + l16;
            const float bias = b1[n];
            const int KC  = n >> 5;
            const int kc  = n & 31;
            const int hf  = (kc >> 3) & 1;
            const int idx = (kc & 7) + ((kc >> 4) << 3);
            #pragma unroll
            for (int r = 0; r < 8; ++r) {
                const int row = wRow + half * 8 + r;
                const float v = acc[t][r] + bias;
                const int lw  = hf * 16 + (row & 15);
                const int strip = row >> 4;
                Tsw[(((KC * NSTRIP + strip) * 32 + lw) << 4) + idx] =
                    to_bf16(v > 0.0f ? v : 0.0f);
            }
        }
    }
    __syncthreads();

    // ---- stage 2: Hout = (T @ W2 + b2) * mask ----
    {
        v8f acc[NT2] = {};
        for (int kc = 0; kc < KC2; ++kc) {
            const v16bf af = *reinterpret_cast<const v16bf*>(
                &Tsw[((kc * NSTRIP + (wave & 3)) * 32 + lane) << 4]);
            #pragma unroll
            for (int t = 0; t < NT2; ++t) {
                const int tIdx = wGrp * (HOUT / 32) + t;
                const v16bf bfv = *reinterpret_cast<const v16bf*>(
                    &W2sw[((kc * (HOUT / 16) + tIdx) * 32 + lane) << 4]);
                acc[t] = __builtin_amdgcn_wmma_f32_16x16x32_bf16(
                    false, af, false, bfv, (short)0, acc[t], false, false);
            }
        }
        #pragma unroll
        for (int t = 0; t < NT2; ++t) {
            const int n = (wGrp * (HOUT / 32) + t) * 16 + l16;
            const float bias = b2[n];
            #pragma unroll
            for (int r = 0; r < 8; ++r) {
                const int row = wRow + half * 8 + r;
                const float m = mask[(size_t)b * NNODES + rowBase + row];
                Hb[(size_t)row * HOUT + n] = (acc[t][r] + bias) * m;
            }
        }
    }
}

// ---------------------------------------------------------------------------
// g[b,c] = max_i H2[b,i,c];  out = g @ Wfc + bfc   (tiny, scalar)
// ---------------------------------------------------------------------------
__global__ __launch_bounds__(1024)
void readout_fc_kernel(const float* __restrict__ H2,   // [B,N,32]
                       const float* __restrict__ Wfc,  // [32,10]
                       const float* __restrict__ bfc,  // [10]
                       float* __restrict__ out)        // [B,10]
{
    __shared__ float g[BATCH][32];
    const int tid = threadIdx.x;
    const int b   = tid >> 5;
    const int c   = tid & 31;

    float m = -3.4e38f;
    const float* p = H2 + (size_t)b * NNODES * 32 + c;
    for (int i = 0; i < NNODES; ++i) m = fmaxf(m, p[(size_t)i * 32]);
    g[b][c] = m;
    __syncthreads();

    if (tid < BATCH * 10) {
        const int bb = tid / 10, o = tid % 10;
        float s = bfc[o];
        #pragma unroll
        for (int cc = 0; cc < 32; ++cc) s += g[bb][cc] * Wfc[cc * 10 + o];
        out[bb * 10 + o] = s;
    }
}

// ---------------------------------------------------------------------------
extern "C" void kernel_launch(void* const* d_in, const int* in_sizes, int n_in,
                              void* d_out, int out_size, void* d_ws, size_t ws_size,
                              hipStream_t stream)
{
    const float* x    = (const float*)d_in[0];
    const float* adj  = (const float*)d_in[1];
    const float* mask = (const float*)d_in[2];
    const float* W1a  = (const float*)d_in[3];
    const float* b1a  = (const float*)d_in[4];
    const float* W2a  = (const float*)d_in[5];
    const float* b2a  = (const float*)d_in[6];
    const float* W1b  = (const float*)d_in[7];
    const float* b1b  = (const float*)d_in[8];
    const float* W2b  = (const float*)d_in[9];
    const float* b2b  = (const float*)d_in[10];
    const float* Wfc  = (const float*)d_in[11];
    const float* bfc  = (const float*)d_in[12];
    float* out = (float*)d_out;

    // workspace layout (f32): Z1[B,N,128] | H1[B,N,64] | Z2[B,N,64] | H2[B,N,32]
    float* Z1 = (float*)d_ws;
    float* H1 = Z1 + (size_t)BATCH * NNODES * 128;
    float* Z2 = H1 + (size_t)BATCH * NNODES * 64;
    float* H2 = Z2 + (size_t)BATCH * NNODES * 64;

    const dim3 blk(256);
    const dim3 grid(NNODES / 64, BATCH);

    agg_gemm_kernel<128><<<grid, blk, 0, stream>>>(adj, x, Z1);
    mlp_gemm_kernel<128, 64, 64><<<grid, blk, 0, stream>>>(Z1, W1a, b1a, W2a, b2a, mask, H1);
    agg_gemm_kernel<64><<<grid, blk, 0, stream>>>(adj, H1, Z2);
    mlp_gemm_kernel<64, 32, 32><<<grid, blk, 0, stream>>>(Z2, W1b, b1b, W2b, b2b, mask, H2);
    readout_fc_kernel<<<dim3(1), dim3(1024), 0, stream>>>(H2, Wfc, bfc, out);
}
